// ConformerConvolutionQuant_68315749810755
// MI455X (gfx1250) — 8-bit matrix operand order repaired and checked by reading, compiled, not run on hardware
//
#include <hip/hip_runtime.h>
#include <stdint.h>

typedef int v8i __attribute__((ext_vector_type(8)));

#define BQ 16
#define TQ 2048
#define CQ 512
#define KQ 31
#define BT (BQ*TQ)
#define TWOC (2*CQ)

#define OFF_A   0ull
#define OFF_H6  67108864ull
#define OFF_B   134217728ull
#define OFF_C   201326592ull
#define OFF_W1P 218103808ull
#define OFF_W2P 218628096ull
#define OFF_S   218890240ull

#define S_AW1 0
#define S_ADW 1
#define S_AW2 2
#define S_A1T 16
#define S_A2T 2064
#define S_A3C 4112
#define S_A4C 4624
#define S_A5T 5136
#define S_A6T 7184
#define S_BNM 9232
#define S_BNV 9744
#define S_TOTAL 10256

__device__ __forceinline__ float q_scale(float amax) {
    return fmaxf(amax, 1e-8f) * (1.0f / 127.0f);
}
__device__ __forceinline__ float fq_val(float x, float s) {
    float q = rintf(x / s);
    q = fminf(fmaxf(q, -128.0f), 127.0f);
    return q * s;
}
__device__ __forceinline__ int q8_val(float x, float s) {
    float q = rintf(x / s);
    q = fminf(fmaxf(q, -128.0f), 127.0f);
    return (int)q;
}
__device__ __forceinline__ float blockReduceMax(float v, float* sm) {
    int tid = threadIdx.x;
    sm[tid] = v; __syncthreads();
    for (int s = 128; s > 0; s >>= 1) {
        if (tid < s) sm[tid] = fmaxf(sm[tid], sm[tid + s]);
        __syncthreads();
    }
    float r = sm[0]; __syncthreads();
    return r;
}
__device__ __forceinline__ float blockReduceSum(float v, float* sm) {
    int tid = threadIdx.x;
    sm[tid] = v; __syncthreads();
    for (int s = 128; s > 0; s >>= 1) {
        if (tid < s) sm[tid] += sm[tid + s];
        __syncthreads();
    }
    float r = sm[0]; __syncthreads();
    return r;
}
__device__ __forceinline__ uint32_t lds_off(const void* p) {
    return (uint32_t)(size_t)p;
}
__device__ __forceinline__ void async_copy_b128(uint32_t lds_dst, const void* gsrc) {
    asm volatile("global_load_async_to_lds_b128 %0, %1, off"
                 :: "v"(lds_dst), "v"(gsrc) : "memory");
}
__device__ __forceinline__ void wait_async_le1() {
    asm volatile("s_wait_asynccnt 0x1" ::: "memory");
}
__device__ __forceinline__ void wait_async_le0() {
    asm volatile("s_wait_asynccnt 0x0" ::: "memory");
}

__global__ void init_stats_kernel(float* s, int n) {
    int i = blockIdx.x * 256 + threadIdx.x;
    if (i < n) s[i] = 0.0f;
}

__global__ void absmax_kernel(const float* __restrict__ x, int n, float* __restrict__ out) {
    __shared__ float sm[256];
    float m = 0.0f;
    for (int i = blockIdx.x * blockDim.x + threadIdx.x; i < n; i += gridDim.x * blockDim.x)
        m = fmaxf(m, fabsf(x[i]));
    m = blockReduceMax(m, sm);
    if (threadIdx.x == 0) atomicMax((unsigned int*)out, __float_as_uint(m));
}

__global__ void pack_w_i8_kernel(const float* __restrict__ W, const float* __restrict__ amaxW,
                                 int8_t* __restrict__ Out, int N, int K) {
    const float s = q_scale(*amaxW);
    const int NT = N >> 4;
    const int total = N * K;
    for (int idx = blockIdx.x * blockDim.x + threadIdx.x; idx < total; idx += gridDim.x * blockDim.x) {
        int tile = idx >> 10;
        int w    = idx & 1023;
        int lane = w >> 5;
        int rem  = w & 31;
        int v    = rem >> 2;
        int byt  = rem & 3;
        int kt = tile / NT;
        int nt = tile - kt * NT;
        int n = nt * 16 + (lane & 15);
        int k = kt * 64 + ((rem >> 3) << 4) + ((lane >> 4) << 3) + (rem & 7);
        Out[idx] = (int8_t)q8_val(W[(size_t)n * K + k], s);
    }
}

__global__ void ln_kernel(const float* __restrict__ x, const float* __restrict__ g,
                          const float* __restrict__ bta, float* __restrict__ h0,
                          float* __restrict__ amax1) {
    __shared__ float sm[256];
    const int row = blockIdx.x;
    const int t = row & (TQ - 1);
    const float* xr = x + (size_t)row * CQ;
    const int i0 = threadIdx.x, i1 = threadIdx.x + 256;
    float x0 = xr[i0], x1 = xr[i1];
    float mu = blockReduceSum(x0 + x1, sm) * (1.0f / CQ);
    float d0 = x0 - mu, d1 = x1 - mu;
    float var = blockReduceSum(d0 * d0 + d1 * d1, sm) * (1.0f / CQ);
    float rstd = rsqrtf(var + 1e-5f);
    float y0 = d0 * rstd * g[i0] + bta[i0];
    float y1 = d1 * rstd * g[i1] + bta[i1];
    h0[(size_t)row * CQ + i0] = y0;
    h0[(size_t)row * CQ + i1] = y1;
    float am = blockReduceMax(fmaxf(fabsf(y0), fabsf(y1)), sm);
    if (threadIdx.x == 0) atomicMax((unsigned int*)(amax1 + t), __float_as_uint(am));
}

__global__ void quant_i8_kernel(const float* __restrict__ in, const float* __restrict__ amax,
                                int8_t* __restrict__ out, int shift, int mask) {
    int idx = blockIdx.x * 256 + threadIdx.x;
    float s = q_scale(amax[(idx >> shift) & mask]);
    out[idx] = (int8_t)q8_val(in[idx], s);
}

__global__ void wmma_gemm_i8_kernel(const int8_t* __restrict__ A, const int8_t* __restrict__ Bp,
                                    const float* __restrict__ bias, const float* __restrict__ amaxA_t,
                                    const float* __restrict__ amaxW, float* __restrict__ Out,
                                    int M, int N, int K) {
    __shared__ __align__(16) int8_t bs[2][4096];
    const int tid = threadIdx.x;
    const int lane = tid & 31;
    const int wave = tid >> 5;
    const int NT = N >> 4;
    const int KT = K >> 6;
    const int NB = N >> 6;
    const int mt = blockIdx.x / NB;
    const int nb = blockIdx.x - mt * NB;
    const int half = lane >> 4;

    const int8_t* Arow = A + (size_t)(mt * 128 + wave * 16 + (lane & 15)) * K + half * 8;
    const int8_t* Bblk = Bp + (((size_t)nb * 4) << 10) + (size_t)tid * 16;

    const uint32_t ldsb[2] = { lds_off(&bs[0][0]) + (uint32_t)tid * 16u,
                               lds_off(&bs[1][0]) + (uint32_t)tid * 16u };

    async_copy_b128(ldsb[0], Bblk);

    v8i acc0 = {0,0,0,0,0,0,0,0};
    v8i acc1 = {0,0,0,0,0,0,0,0};
    v8i acc2 = {0,0,0,0,0,0,0,0};
    v8i acc3 = {0,0,0,0,0,0,0,0};

    int cur = 0;
    for (int kt = 0; kt < KT; ++kt) {
        if (kt + 1 < KT) {
            async_copy_b128(ldsb[cur ^ 1], Bblk + (((size_t)(kt + 1) * NT) << 10));
            wait_async_le1();
        } else {
            wait_async_le0();
        }
        __syncthreads();

        union { v8i v; int2 p[4]; } ua;
        const int k0 = kt << 6;
        ua.p[0] = *(const int2*)(Arow + k0);
        ua.p[1] = *(const int2*)(Arow + k0 + 16);
        ua.p[2] = *(const int2*)(Arow + k0 + 32);
        ua.p[3] = *(const int2*)(Arow + k0 + 48);

        const int8_t* bb = &bs[cur][lane << 5];
        v8i b0 = *(const v8i*)(bb);
        v8i b1 = *(const v8i*)(bb + 1024);
        v8i b2 = *(const v8i*)(bb + 2048);
        v8i b3 = *(const v8i*)(bb + 3072);
        acc0 = __builtin_amdgcn_wmma_i32_16x16x64_iu8(true, ua.v, true, b0, acc0, false, false);
        acc1 = __builtin_amdgcn_wmma_i32_16x16x64_iu8(true, ua.v, true, b1, acc1, false, false);
        acc2 = __builtin_amdgcn_wmma_i32_16x16x64_iu8(true, ua.v, true, b2, acc2, false, false);
        acc3 = __builtin_amdgcn_wmma_i32_16x16x64_iu8(true, ua.v, true, b3, acc3, false, false);

        __syncthreads();
        cur ^= 1;
    }

    const float sw = q_scale(*amaxW);
    float sa[8];
#pragma unroll
    for (int r = 0; r < 8; ++r)
        sa[r] = q_scale(amaxA_t[(mt * 128 + wave * 16 + r + half * 8) & (TQ - 1)]);

    v8i accs[4] = {acc0, acc1, acc2, acc3};
#pragma unroll
    for (int j = 0; j < 4; ++j) {
        const int n = nb * 64 + j * 16 + (lane & 15);
        const float bn = bias[n];
#pragma unroll
        for (int r = 0; r < 8; ++r) {
            const int m = mt * 128 + wave * 16 + r + half * 8;
            Out[(size_t)m * N + n] = (float)accs[j][r] * (sa[r] * sw) + bn;
        }
    }
}

__global__ void amax2_kernel(const float* __restrict__ h1, float* __restrict__ out) {
    __shared__ float sm[256];
    const int t = blockIdx.x;
    float m = 0.0f;
    for (int b = 0; b < BQ; ++b) {
        const float* row = h1 + (size_t)(b * TQ + t) * TWOC;
        for (int i = threadIdx.x; i < TWOC; i += 256) m = fmaxf(m, fabsf(row[i]));
    }
    m = blockReduceMax(m, sm);
    if (threadIdx.x == 0) out[t] = m;
}

__global__ void glu_kernel(const float* __restrict__ h1, const float* __restrict__ amax2,
                           float* __restrict__ glu) {
    int idx = blockIdx.x * 256 + threadIdx.x;
    int t = idx & (TQ - 1);
    int c = (idx >> 11) & (CQ - 1);
    int b = idx >> 20;
    float s2 = q_scale(amax2[t]);
    size_t row = (size_t)(b * TQ + t) * TWOC;
    float a = fq_val(h1[row + c], s2);
    float g = fq_val(h1[row + c + CQ], s2);
    glu[idx] = a / (1.0f + expf(-g));
}

__global__ void amax_c_kernel(const float* __restrict__ x, float* __restrict__ out) {
    __shared__ float sm[256];
    const int c = blockIdx.x;
    float m = 0.0f;
    for (int b = 0; b < BQ; ++b) {
        const float* row = x + (size_t)(b * CQ + c) * TQ;
        for (int i = threadIdx.x; i < TQ; i += 256) m = fmaxf(m, fabsf(row[i]));
    }
    m = blockReduceMax(m, sm);
    if (threadIdx.x == 0) out[c] = m;
}

__global__ void dwconv_kernel(const int8_t* __restrict__ gluq, const float* __restrict__ dw,
                              const float* __restrict__ db, const float* __restrict__ amax3,
                              const float* __restrict__ amaxDW, float* __restrict__ out) {
    __shared__ __align__(16) int8_t srow[16 + TQ + 16];
    const int tid = threadIdx.x;
    const int bc = blockIdx.x;
    const int c = bc & (CQ - 1);
    const int8_t* in = gluq + (size_t)bc * TQ;
    if (tid < 16) {
        srow[tid] = 0;
        srow[16 + TQ + tid] = 0;
    }
    if (tid < TQ / 16) {
        async_copy_b128(lds_off(&srow[16]) + (uint32_t)tid * 16u, in + tid * 16);
        wait_async_le0();
    }
    const float sdw = q_scale(*amaxDW);
    int qd[KQ];
#pragma unroll
    for (int k = 0; k < KQ; ++k) qd[k] = q8_val(dw[c * KQ + k], sdw);
    __syncthreads();
    const float s = q_scale(amax3[c]) * sdw;
    const float bias = db[c];
    for (int t = tid; t < TQ; t += 256) {
        int acc = 0;
#pragma unroll
        for (int k = 0; k < KQ; ++k) acc += (int)srow[t + k + 1] * qd[k];
        out[(size_t)bc * TQ + t] = (float)acc * s + bias;
    }
}

__global__ void bnstats_kernel(const float* __restrict__ conv, const float* __restrict__ amax4,
                               float* __restrict__ bnm, float* __restrict__ bnv) {
    __shared__ float sm[256];
    const int c = blockIdx.x;
    const float s4 = q_scale(amax4[c]);
    float sum = 0.0f, sq = 0.0f;
    for (int b = 0; b < BQ; ++b) {
        const float* row = conv + (size_t)(b * CQ + c) * TQ;
        for (int i = threadIdx.x; i < TQ; i += 256) {
            float v = fq_val(row[i], s4);
            sum += v; sq += v * v;
        }
    }
    float S = blockReduceSum(sum, sm);
    float Q = blockReduceSum(sq, sm);
    if (threadIdx.x == 0) {
        float mu = S * (1.0f / (BQ * TQ));
        bnm[c] = mu;
        bnv[c] = Q * (1.0f / (BQ * TQ)) - mu * mu;
    }
}

__global__ void bnsilu_kernel(const float* __restrict__ conv, const float* __restrict__ amax4,
                              const float* __restrict__ bnm, const float* __restrict__ bnv,
                              const float* __restrict__ gamma, const float* __restrict__ beta,
                              float* __restrict__ h5, float* __restrict__ amax5) {
    __shared__ float sm[256];
    const int row = blockIdx.x;
    const int b = row >> 11;
    const int t = row & (TQ - 1);
    float am = 0.0f;
#pragma unroll
    for (int j = 0; j < 2; ++j) {
        int c = threadIdx.x + j * 256;
        float s4 = q_scale(amax4[c]);
        float v = fq_val(conv[(size_t)(b * CQ + c) * TQ + t], s4);
        v = (v - bnm[c]) * rsqrtf(bnv[c] + 1e-5f) * gamma[c] + beta[c];
        v = v / (1.0f + expf(-v));
        h5[(size_t)row * CQ + c] = v;
        am = fmaxf(am, fabsf(v));
    }
    am = blockReduceMax(am, sm);
    if (threadIdx.x == 0) atomicMax((unsigned int*)(amax5 + t), __float_as_uint(am));
}

__global__ void amax6_kernel(const float* __restrict__ h6, float* __restrict__ out) {
    __shared__ float sm[256];
    const int t = blockIdx.x;
    float m = 0.0f;
    for (int b = 0; b < BQ; ++b) {
        const float* row = h6 + (size_t)(b * TQ + t) * CQ;
        for (int i = threadIdx.x; i < CQ; i += 256) m = fmaxf(m, fabsf(row[i]));
    }
    m = blockReduceMax(m, sm);
    if (threadIdx.x == 0) out[t] = m;
}

__global__ void final_quant_kernel(const float* __restrict__ h6, const float* __restrict__ amax6,
                                   float* __restrict__ out) {
    int idx = blockIdx.x * 256 + threadIdx.x;
    float s = q_scale(amax6[(idx >> 9) & (TQ - 1)]);
    out[idx] = fq_val(h6[idx], s);
}

extern "C" void kernel_launch(void* const* d_in, const int* in_sizes, int n_in,
                              void* d_out, int out_size, void* d_ws, size_t ws_size,
                              hipStream_t stream) {
    const float* x        = (const float*)d_in[0];
    const float* ln_gamma = (const float*)d_in[1];
    const float* ln_beta  = (const float*)d_in[2];
    const float* w1       = (const float*)d_in[3];
    const float* b1       = (const float*)d_in[4];
    const float* dw       = (const float*)d_in[5];
    const float* db       = (const float*)d_in[6];
    const float* bn_gamma = (const float*)d_in[7];
    const float* bn_beta  = (const float*)d_in[8];
    const float* w2       = (const float*)d_in[9];
    const float* b2       = (const float*)d_in[10];
    float* out = (float*)d_out;

    char* ws = (char*)d_ws;
    float*  h1   = (float*)(ws + OFF_A);
    float*  conv = (float*)(ws + OFF_A);
    float*  h6   = (float*)(ws + OFF_H6);
    float*  h0   = (float*)(ws + OFF_B);
    float*  glu  = (float*)(ws + OFF_B);
    float*  h5   = (float*)(ws + OFF_B);
    int8_t* q8a  = (int8_t*)(ws + OFF_C);
    int8_t* w1p  = (int8_t*)(ws + OFF_W1P);
    int8_t* w2p  = (int8_t*)(ws + OFF_W2P);
    float*  st   = (float*)(ws + OFF_S);

    init_stats_kernel<<<(S_TOTAL + 255) / 256, 256, 0, stream>>>(st, S_TOTAL);

    absmax_kernel<<<512, 256, 0, stream>>>(w1, TWOC * CQ, st + S_AW1);
    absmax_kernel<<<62,  256, 0, stream>>>(dw, CQ * KQ,   st + S_ADW);
    absmax_kernel<<<512, 256, 0, stream>>>(w2, CQ * CQ,   st + S_AW2);

    pack_w_i8_kernel<<<2048, 256, 0, stream>>>(w1, st + S_AW1, w1p, TWOC, CQ);
    pack_w_i8_kernel<<<1024, 256, 0, stream>>>(w2, st + S_AW2, w2p, CQ, CQ);

    ln_kernel<<<BT, 256, 0, stream>>>(x, ln_gamma, ln_beta, h0, st + S_A1T);

    quant_i8_kernel<<<(BT * CQ) / 256, 256, 0, stream>>>(h0, st + S_A1T, q8a, 9, TQ - 1);

    wmma_gemm_i8_kernel<<<(BT / 128) * (TWOC / 64), 256, 0, stream>>>(
        q8a, w1p, b1, st + S_A1T, st + S_AW1, h1, BT, TWOC, CQ);

    amax2_kernel<<<TQ, 256, 0, stream>>>(h1, st + S_A2T);
    glu_kernel<<<(BQ * CQ * TQ) / 256, 256, 0, stream>>>(h1, st + S_A2T, glu);

    amax_c_kernel<<<CQ, 256, 0, stream>>>(glu, st + S_A3C);
    quant_i8_kernel<<<(BQ * CQ * TQ) / 256, 256, 0, stream>>>(glu, st + S_A3C, q8a, 11, CQ - 1);

    dwconv_kernel<<<BQ * CQ, 256, 0, stream>>>(q8a, dw, db, st + S_A3C, st + S_ADW, conv);

    amax_c_kernel<<<CQ, 256, 0, stream>>>(conv, st + S_A4C);
    bnstats_kernel<<<CQ, 256, 0, stream>>>(conv, st + S_A4C, st + S_BNM, st + S_BNV);

    bnsilu_kernel<<<BT, 256, 0, stream>>>(conv, st + S_A4C, st + S_BNM, st + S_BNV,
                                          bn_gamma, bn_beta, h5, st + S_A5T);

    quant_i8_kernel<<<(BT * CQ) / 256, 256, 0, stream>>>(h5, st + S_A5T, q8a, 9, TQ - 1);

    wmma_gemm_i8_kernel<<<(BT / 128) * (CQ / 64), 256, 0, stream>>>(
        q8a, w2p, b2, st + S_A5T, st + S_AW2, h6, BT, CQ, CQ);

    amax6_kernel<<<TQ, 256, 0, stream>>>(h6, st + S_A6T);
    final_quant_kernel<<<(BT * CQ) / 256, 256, 0, stream>>>(h6, st + S_A6T, out);

    (void)in_sizes; (void)n_in; (void)out_size; (void)ws_size;
}
